// TCFormerDynamicAttention_28063316312345
// MI455X (gfx1250) — matrix operand order repaired and checked by reading, not compiled, not run on hardware
//
#include <hip/hip_runtime.h>
#include <hip/hip_bf16.h>

// ---------------------------------------------------------------------------
// TCFormer dynamic attention, CDNA5 (gfx1250), wave32, WMMA bf16 pipeline.
// Shapes fixed by harness: B=8, Nq=Nkv=12544, C=128, 2 heads x 64.
// All WMMA fragments are loaded as pairs of 16-byte (b128) loads from
// fragment-friendly layouts so operands land as contiguous VGPR quads.
// ---------------------------------------------------------------------------

#define B_      8
#define NQ_     12544
#define C_      128
#define NH_     2
#define HD_     64
#define H_      112
#define W_      112
#define HO_     28
#define WO_     28
#define NS_     784        // 28*28 reduced keys
#define NSP_    896        // padded to 7*128
#define KT_     128        // key tile
#define NKT_    7          // NSP_/KT_
#define SCALE_  0.125f     // 64^-0.5
#define CC_     129        // C + conf channel

typedef __attribute__((ext_vector_type(16))) __bf16    v16bf;
typedef __attribute__((ext_vector_type(8)))  float     v8f;
typedef __attribute__((ext_vector_type(8)))  unsigned  v8u;
typedef __attribute__((ext_vector_type(4)))  unsigned  v4u;

static __device__ __forceinline__ unsigned short bf16bits(float f) {
  unsigned u = __builtin_bit_cast(unsigned, f);
  unsigned r = u + 0x7FFFu + ((u >> 16) & 1u);   // round-to-nearest-even
  return (unsigned short)(r >> 16);
}
static __device__ __forceinline__ unsigned pack2(float a, float b) {
  return (unsigned)bf16bits(a) | ((unsigned)bf16bits(b) << 16);
}
// a0/a1, b0/b1: low/high quads of the A and B fragments (contiguous loads).
static __device__ __forceinline__ v8f wmma_bf16(v4u a0, v4u a1, v4u b0, v4u b1,
                                                v8f c) {
  v8u av = __builtin_shufflevector(a0, a1, 0, 1, 2, 3, 4, 5, 6, 7);
  v8u bv = __builtin_shufflevector(b0, b1, 0, 1, 2, 3, 4, 5, 6, 7);
  return __builtin_amdgcn_wmma_f32_16x16x32_bf16(
      false, __builtin_bit_cast(v16bf, av),
      false, __builtin_bit_cast(v16bf, bv),
      (short)0, c, false, false);
}

// ---------------------------------------------------------------------------
// K1: token2map. Each (b,y,x) map cell averages 4 gathered tokens (count==4).
// map layout: [B][112][112][129] f32, channel-last.
// ---------------------------------------------------------------------------
__global__ __launch_bounds__(128) void k_token2map(
    const float* __restrict__ kv_x, const float* __restrict__ tscore,
    const int* __restrict__ idx_token, float* __restrict__ map) {
  int blk = blockIdx.x;                 // b*12544 + y*112 + x
  int b = blk / (H_ * W_);
  int hw = blk - b * (H_ * W_);
  int y = hw / W_, x = hw - y * W_;
  int ib = b * (2 * H_ * 2 * W_) + (2 * y) * (2 * W_) + 2 * x;
  int t00 = idx_token[ib];
  int t01 = idx_token[ib + 1];
  int t10 = idx_token[ib + 2 * W_];
  int t11 = idx_token[ib + 2 * W_ + 1];
  const float inv = 1.0f / (4.0f + 1e-6f);
  size_t base = (size_t)blk * CC_;
  int c = threadIdx.x;
  size_t tb = (size_t)b * NQ_;
  float v = kv_x[(tb + t00) * C_ + c] + kv_x[(tb + t01) * C_ + c] +
            kv_x[(tb + t10) * C_ + c] + kv_x[(tb + t11) * C_ + c];
  map[base + c] = v * inv;
  if (c == 0) {
    float s = tscore[tb + t00] + tscore[tb + t01] +
              tscore[tb + t10] + tscore[tb + t11];
    map[base + C_] = s * inv;
  }
}

// ---------------------------------------------------------------------------
// K2: 4x4/stride-4 conv + bias + LayerNorm over C + conf 4x4 avg pool.
// ---------------------------------------------------------------------------
__global__ __launch_bounds__(128) void k_conv_ln(
    const float* __restrict__ map, const float* __restrict__ srw,
    const float* __restrict__ srb, const float* __restrict__ ng,
    const float* __restrict__ nb, float* __restrict__ kvt,
    float* __restrict__ conf) {
  __shared__ float patch[16 * CC_];
  __shared__ float red[128];
  int blk = blockIdx.x;
  int b = blk / NS_, n = blk - b * NS_;
  int oh = n / WO_, ow = n - oh * WO_;
  int tid = threadIdx.x;
  for (int i = tid; i < 16 * CC_; i += 128) {
    int p = i / CC_, c = i - p * CC_;
    int kh = p >> 2, kw = p & 3;
    patch[i] = map[((size_t)b * (H_ * W_) +
                    (4 * oh + kh) * W_ + (4 * ow + kw)) * CC_ + c];
  }
  __syncthreads();
  int oc = tid;
  float acc = srb[oc];
  for (int ic = 0; ic < C_; ++ic) {
#pragma unroll
    for (int p = 0; p < 16; ++p)
      acc += patch[p * CC_ + ic] * srw[oc * 2048 + ic * 16 + p];
  }
  red[tid] = acc; __syncthreads();
  for (int s = 64; s > 0; s >>= 1) {
    if (tid < s) red[tid] += red[tid + s];
    __syncthreads();
  }
  float mu = red[0] * (1.0f / 128.0f);
  __syncthreads();
  float d = acc - mu;
  red[tid] = d * d; __syncthreads();
  for (int s = 64; s > 0; s >>= 1) {
    if (tid < s) red[tid] += red[tid + s];
    __syncthreads();
  }
  float var = red[0] * (1.0f / 128.0f);
  float val = d * rsqrtf(var + 1e-5f) * ng[oc] + nb[oc];
  kvt[((size_t)b * NS_ + n) * C_ + oc] = val;
  if (tid == 0) {
    float cs = 0.f;
#pragma unroll
    for (int p = 0; p < 16; ++p) cs += patch[p * CC_ + C_];
    conf[b * NSP_ + n] = cs * (1.0f / 16.0f);
    if (n < NSP_ - NS_) conf[b * NSP_ + NS_ + n] = -1e30f;  // pad-key mask
  }
}

// ---------------------------------------------------------------------------
// K3: KV projection producing WMMA-fragment-friendly layouts:
//   Kp [bh][key n (896)][32 d-pairs]  u32  (bf16 pair along d; zero pad)
//   Vp [bh][d (64)][448 key-pairs]    u32  (bf16 pair along key; zero pad)
// A lane's B-fragment (8 dwords) is two b128 quads, 2 quads apart, in both layouts.
// ---------------------------------------------------------------------------
__global__ __launch_bounds__(128) void k_kvproj(
    const float* __restrict__ kvt, const float* __restrict__ kvw,
    unsigned* __restrict__ Kp, unsigned* __restrict__ Vp) {
  int gid = blockIdx.x * 128 + threadIdx.x;
  if (blockIdx.y == 0) {                       // K path
    int r = gid & 31;
    int n = (gid >> 5) % NSP_;
    int bh = gid / (32 * NSP_);
    int b = bh >> 1, head = bh & 1;
    size_t op = ((size_t)bh * NSP_ + n) * 32 + r;
    if (n >= NS_) { Kp[op] = 0u; return; }
    const float* a = kvt + ((size_t)b * NS_ + n) * C_;
    int c0 = head * HD_ + 2 * r;
    float s0 = 0.f, s1 = 0.f;
    for (int k = 0; k < C_; ++k) {
      float av = a[k];
      s0 += av * kvw[k * 256 + c0];
      s1 += av * kvw[k * 256 + c0 + 1];
    }
    Kp[op] = pack2(s0, s1);
  } else {                                     // V path
    int d = gid & 63;
    int s = (gid >> 6) % 448;
    int bh = gid / (64 * 448);
    int b = bh >> 1, head = bh & 1;
    size_t op = ((size_t)bh * HD_ + d) * 448 + s;
    int n0 = 2 * s;
    if (n0 >= NS_) { Vp[op] = 0u; return; }
    int col = 2 * HD_ + head * HD_ + d;        // v half of kv_w
    const float* a0 = kvt + ((size_t)b * NS_ + n0) * C_;
    float s0 = 0.f, s1 = 0.f;
    for (int k = 0; k < C_; ++k) {
      float w = kvw[k * 256 + col];
      s0 += a0[k] * w;
      s1 += a0[C_ + k] * w;
    }
    Vp[op] = pack2(s0, s1);
  }
}

// ---------------------------------------------------------------------------
// Shared WMMA GEMM body for the 100352x128x128 projections.
// Wn: [n][k-pair] packed bf16 weights in LDS; Ap: [row][k-pair].
// ---------------------------------------------------------------------------
#define GEMM_BODY(ACC)                                                        \
  int wave = tid >> 5, lane = tid & 31;                                       \
  int cl = lane & 15, half = lane >> 4;                                       \
  const v4u* Av = (const v4u*)Ap + (wave * 16 + cl) * 16;                     \
  const v4u* Wv = (const v4u*)Wn;                                             \
  v8f ACC[8] = {};                                                            \
  for (int s = 0; s < 4; ++s) {                                               \
    v4u a0 = Av[4 * s + half];                                                \
    v4u a1 = Av[4 * s + half + 2];                                            \
    _Pragma("unroll")                                                         \
    for (int t = 0; t < 8; ++t) {                                             \
      const v4u* wc = Wv + (16 * t + cl) * 16 + 4 * s + half;                 \
      ACC[t] = wmma_bf16(a0, a1, wc[0], wc[2], ACC[t]);                       \
    }                                                                         \
  }

// K4: Q projection — WMMA bf16 GEMM, SCALE folded in, bf16 row-major output.
__global__ __launch_bounds__(128) void k_qproj(
    const float* __restrict__ qx, const float* __restrict__ qw,
    unsigned short* __restrict__ Qb) {
  __shared__ __align__(16) unsigned Wn[128 * 64];  // [n][k-pair]
  __shared__ __align__(16) unsigned Ap[64 * 64];   // [row][k-pair]
  int tid = threadIdx.x;
  size_t row0 = (size_t)blockIdx.x * 64;
  for (int i = tid; i < 128 * 64; i += 128) {
    int pr = i & 63, nn = i >> 6;
    Wn[nn * 64 + pr] = pack2(qw[(2 * pr) * C_ + nn], qw[(2 * pr + 1) * C_ + nn]);
  }
  for (int i = tid; i < 64 * 64; i += 128) {
    int m = i >> 6, p = i & 63;
    const float* a = qx + (row0 + m) * C_ + 2 * p;
    Ap[i] = pack2(a[0], a[1]);
  }
  __syncthreads();
  GEMM_BODY(acc)
#pragma unroll
  for (int t = 0; t < 8; ++t)
#pragma unroll
    for (int r = 0; r < 8; ++r) {
      size_t row = row0 + wave * 16 + half * 8 + r;
      Qb[row * C_ + 16 * t + cl] = bf16bits(acc[t][r] * SCALE_);
    }
}

// K6: output projection — WMMA bf16 GEMM + bias, f32 result.
__global__ __launch_bounds__(128) void k_proj(
    const unsigned short* __restrict__ Ob, const float* __restrict__ pw,
    const float* __restrict__ pb, float* __restrict__ out) {
  __shared__ __align__(16) unsigned Wn[128 * 64];
  __shared__ __align__(16) unsigned Ap[64 * 64];
  int tid = threadIdx.x;
  size_t row0 = (size_t)blockIdx.x * 64;
  for (int i = tid; i < 128 * 64; i += 128) {
    int pr = i & 63, nn = i >> 6;
    Wn[nn * 64 + pr] = pack2(pw[(2 * pr) * C_ + nn], pw[(2 * pr + 1) * C_ + nn]);
  }
  const v4u* Ou = (const v4u*)Ob;
  v4u* Ap4 = (v4u*)Ap;
  for (int i = tid; i < 64 * 16; i += 128)     // 64 rows x 16 v4u
    Ap4[i] = Ou[(row0 + (i >> 4)) * 16 + (i & 15)];
  __syncthreads();
  GEMM_BODY(acc)
#pragma unroll
  for (int t = 0; t < 8; ++t)
#pragma unroll
    for (int r = 0; r < 8; ++r) {
      size_t row = row0 + wave * 16 + half * 8 + r;
      int col = 16 * t + cl;
      out[row * C_ + col] = acc[t][r] + pb[col];
    }
}

// ---------------------------------------------------------------------------
// K5: fused flash attention. Block = 4 waves x 16 query rows, one (b,head).
// lK: [key][d-pair] (16 KB, linear copy of a Kp slab); lV: [d][key-pair].
// Every WMMA operand fragment = two contiguous b128 loads.
// ---------------------------------------------------------------------------
__global__ __launch_bounds__(128) void k_attn(
    const unsigned short* __restrict__ Qb, const unsigned* __restrict__ Kp,
    const unsigned* __restrict__ Vp, const float* __restrict__ conf,
    unsigned short* __restrict__ Ob) {
  __shared__ __align__(16) unsigned lK[KT_ * 32];          // 16 KB
  __shared__ __align__(16) unsigned lV[HD_ * 64];          // 16 KB
  __shared__ __align__(16) float lc[KT_];
  __shared__ __align__(16) unsigned short lP[4][16 * KT_]; // 16 KB
  int bh = blockIdx.x, b = bh >> 1, head = bh & 1;
  int tid = threadIdx.x, wave = tid >> 5, lane = tid & 31;
  int cl = lane & 15, half = lane >> 4;
  int q0 = blockIdx.y * 64 + wave * 16;      // wave's query row base
  size_t qrow = (size_t)b * NQ_ + q0 + cl;

  // Q A-fragments: two b128 global loads per K32 step.
  const v4u* Qv = (const v4u*)((const unsigned*)Qb + qrow * 64 + head * 32);
  v4u qa[2][2];
#pragma unroll
  for (int s = 0; s < 2; ++s) {
    qa[s][0] = Qv[4 * s + half];
    qa[s][1] = Qv[4 * s + half + 2];
  }

  float mst[8], lst[8];
#pragma unroll
  for (int r = 0; r < 8; ++r) { mst[r] = -3.0e38f; lst[r] = 0.f; }
  v8f o[4] = {};

  for (int kt = 0; kt < NKT_; ++kt) {
    int key0 = kt * KT_;
    // K tile: pure linear 16 KB copy (coalesced b128 in+out).
    const v4u* Kg = (const v4u*)(Kp + ((size_t)bh * NSP_ + key0) * 32);
    v4u* lK4 = (v4u*)lK;
    for (int i = tid; i < KT_ * 8; i += 128) lK4[i] = Kg[i];
    // V tile: 64 rows of 16 v4u, global row stride 112 v4u.
    const v4u* Vg = (const v4u*)Vp;
    v4u* lV4 = (v4u*)lV;
    for (int i = tid; i < HD_ * 16; i += 128)
      lV4[i] = Vg[((size_t)bh * HD_ + (i >> 4)) * 112 + kt * 16 + (i & 15)];
    lc[tid] = conf[b * NSP_ + key0 + tid];
    if (kt + 1 < NKT_) {  // warm caches for the next key tile
      __builtin_prefetch(&Kp[((size_t)bh * NSP_ + key0 + KT_) * 32], 0, 1);
      __builtin_prefetch(&Vp[((size_t)bh * HD_) * 448 + (kt + 1) * 64], 0, 1);
    }
    __syncthreads();

    // S = Q K^T + conf  (8 n-tiles of 16 keys); conf folded into C init.
    v8f sf[8];
#pragma unroll
    for (int t = 0; t < 8; ++t) {
      float cv = lc[16 * t + cl];
      v8f c = {cv, cv, cv, cv, cv, cv, cv, cv};
      const v4u* kc = (const v4u*)lK + (16 * t + cl) * 8;
#pragma unroll
      for (int s = 0; s < 2; ++s)
        c = wmma_bf16(qa[s][0], qa[s][1], kc[4 * s + half],
                      kc[4 * s + half + 2], c);
      sf[t] = c;
    }

    // online softmax: row max/sum via 16-lane butterflies (rows = half*8+r)
    float scale[8];
#pragma unroll
    for (int r = 0; r < 8; ++r) {
      float v = sf[0][r];
#pragma unroll
      for (int t = 1; t < 8; ++t) v = fmaxf(v, sf[t][r]);
      for (int mask = 1; mask < 16; mask <<= 1)
        v = fmaxf(v, __shfl_xor(v, mask, 32));
      float mn = fmaxf(mst[r], v);
      scale[r] = __expf(mst[r] - mn);
      mst[r] = mn;
    }
    float rs[8] = {0.f, 0.f, 0.f, 0.f, 0.f, 0.f, 0.f, 0.f};
#pragma unroll
    for (int t = 0; t < 8; ++t)
#pragma unroll
      for (int r = 0; r < 8; ++r) {
        float p = __expf(sf[t][r] - mst[r]);
        sf[t][r] = p;
        rs[r] += p;
      }
#pragma unroll
    for (int r = 0; r < 8; ++r) {
      float v = rs[r];
      for (int mask = 1; mask < 16; mask <<= 1) v += __shfl_xor(v, mask, 32);
      lst[r] = lst[r] * scale[r] + v;
    }
#pragma unroll
    for (int dt = 0; dt < 4; ++dt)
#pragma unroll
      for (int r = 0; r < 8; ++r) o[dt][r] *= scale[r];

    // stage P (bf16) per-wave, then PV via WMMA (4 d-tiles x 4 K32 steps)
#pragma unroll
    for (int t = 0; t < 8; ++t)
#pragma unroll
      for (int r = 0; r < 8; ++r)
        lP[wave][(half * 8 + r) * KT_ + 16 * t + cl] = bf16bits(sf[t][r]);
    const v4u* Pv = (const v4u*)&lP[wave][0] + cl * 16;
    v4u pa[4][2];
#pragma unroll
    for (int s = 0; s < 4; ++s) {
      pa[s][0] = Pv[4 * s + half];
      pa[s][1] = Pv[4 * s + half + 2];
    }
#pragma unroll
    for (int dt = 0; dt < 4; ++dt) {
      v8f c = o[dt];
      const v4u* vc = (const v4u*)lV + (16 * dt + cl) * 16;
#pragma unroll
      for (int s = 0; s < 4; ++s)
        c = wmma_bf16(pa[s][0], pa[s][1], vc[4 * s + half],
                      vc[4 * s + half + 2], c);
      o[dt] = c;
    }
    __syncthreads();
  }

  // normalize and emit bf16 row-major [B*Nq][128] (head slice)
#pragma unroll
  for (int dt = 0; dt < 4; ++dt)
#pragma unroll
    for (int r = 0; r < 8; ++r) {
      size_t row = (size_t)b * NQ_ + q0 + half * 8 + r;
      int col = head * HD_ + 16 * dt + cl;
      Ob[row * C_ + col] = bf16bits(o[dt][r] / lst[r]);
    }
}

// ---------------------------------------------------------------------------
// Host launcher
// ---------------------------------------------------------------------------
extern "C" void kernel_launch(void* const* d_in, const int* in_sizes, int n_in,
                              void* d_out, int out_size, void* d_ws,
                              size_t ws_size, hipStream_t stream) {
  const float* q_x    = (const float*)d_in[0];
  const float* kv_x   = (const float*)d_in[1];
  const float* tscore = (const float*)d_in[2];
  const int*   idxt   = (const int*)d_in[3];
  const float* q_w    = (const float*)d_in[8];
  const float* kv_w   = (const float*)d_in[9];
  const float* proj_w = (const float*)d_in[10];
  const float* proj_b = (const float*)d_in[11];
  const float* sr_w   = (const float*)d_in[12];
  const float* sr_b   = (const float*)d_in[13];
  const float* norm_g = (const float*)d_in[14];
  const float* norm_b = (const float*)d_in[15];
  float* out = (float*)d_out;

  char* ws = (char*)d_ws;
  // map region: B*H*W*129*4 = 51,781,632 bytes; later reused for Qb/Ob (bf16)
  float* map = (float*)ws;
  unsigned short* Qb = (unsigned short*)ws;                   // 25,690,112 B
  unsigned short* Ob = (unsigned short*)(ws + 25690112);      // 25,690,112 B
  size_t off = 51781632;
  float* kv_t = (float*)(ws + off); off += (size_t)B_ * NS_ * C_ * 4;
  float* conf = (float*)(ws + off); off += (size_t)B_ * NSP_ * 4;
  unsigned* Kp = (unsigned*)(ws + off); off += (size_t)B_ * NH_ * NSP_ * 32 * 4;
  unsigned* Vp = (unsigned*)(ws + off); off += (size_t)B_ * NH_ * HD_ * 448 * 4;

  k_token2map<<<dim3(B_ * H_ * W_), 128, 0, stream>>>(kv_x, tscore, idxt, map);
  k_conv_ln  <<<dim3(B_ * NS_),     128, 0, stream>>>(map, sr_w, sr_b, norm_g,
                                                      norm_b, kv_t, conf);
  k_kvproj   <<<dim3(3584, 2),      128, 0, stream>>>(kv_t, kv_w, Kp, Vp);
  // Qb aliases the map region -> must launch after k_conv_ln consumed map.
  k_qproj    <<<dim3((B_ * NQ_) / 64), 128, 0, stream>>>(q_x, q_w, Qb);
  k_attn     <<<dim3(B_ * NH_, NQ_ / 64), 128, 0, stream>>>(Qb, Kp, Vp, conf, Ob);
  k_proj     <<<dim3((B_ * NQ_) / 64), 128, 0, stream>>>(Ob, proj_w, proj_b, out);
}
